// CausalWanSelfAttention_53798760349902
// MI455X (gfx1250) — compile-verified, matrix operand order checked by reading, not run on hardware
//
#include <hip/hip_runtime.h>
#include <hip/hip_bf16.h>
#include <math.h>

// Problem constants (from reference)
#define L_TOT   3683            // FIRST + IMGB + ALEN + SLEN
#define L16     3696            // padded to 16  (231 tiles)
#define LPAD    3712            // padded to 64  (58 tiles) for 64-row GEMM blocking
#define NTILE_M 231
#define NT64_M  58
#define DIM     1536
#define NH      12
#define HD      128
#define EPSV    1e-6f

typedef __bf16 bf16;
typedef __attribute__((ext_vector_type(16))) __bf16 v16bf;
typedef __attribute__((ext_vector_type(8)))  __bf16 v8bf;
typedef __attribute__((ext_vector_type(8)))  float  v8f;
typedef __attribute__((__vector_size__(16))) int    v4i;   // matches async builtin param

union BFPack { uint4 u4; v8bf v; };

// Load a 16x16x32 WMMA A/B fragment when the operand's K axis is contiguous
// in memory per lane-row. ISA layout (CDNA5 ISA): lanes 0-15 hold K {0..7,16..23},
// lanes 16-31 hold K {8..15,24..31}; each VGPR packs a contiguous K pair.
__device__ __forceinline__ v16bf load_frag(const bf16* rowptr, int k0, int khalf) {
  BFPack a, b;
  a.u4 = *(const uint4*)(rowptr + k0 + 8 * khalf);        // K = 0..7   (+8*khalf)
  b.u4 = *(const uint4*)(rowptr + k0 + 16 + 8 * khalf);   // K = 16..23 (+8*khalf)
  v16bf f;
#pragma unroll
  for (int i = 0; i < 8; ++i) { f[i] = a.v[i]; f[i + 8] = b.v[i]; }
  return f;
}

__device__ __forceinline__ v8f wmma_bf16(v16bf a, v16bf b, v8f c) {
  return __builtin_amdgcn_wmma_f32_16x16x32_bf16(false, a, false, b, (short)0, c,
                                                 false, false);
}

// ---------------------------------------------------------------- converts
__global__ void k_convert_x(const float* __restrict__ x, bf16* __restrict__ xb) {
  int idx = blockIdx.x * 256 + threadIdx.x;               // LPAD*DIM exact
  float v = (idx < L_TOT * DIM) ? x[idx] : 0.0f;          // zero the pad rows
  xb[idx] = (bf16)v;
}

__global__ void k_transpose_w(const float* __restrict__ W, bf16* __restrict__ wt) {
  int idx = blockIdx.x * 256 + threadIdx.x;               // DIM*DIM exact
  int n = idx / DIM, k = idx - n * DIM;
  wt[idx] = (bf16)W[(size_t)k * DIM + n];                 // wt[n][k] = W[k][n]
}

// -------------------------------------------------- 64x32 register-blocked GEMM
// C(LPAD x DIM) = A(LPAD x DIM) * WT^T + bias ; one wave per 64x32 output tile:
// 8 WMMAs per 12 b128 fragment loads per K-step. amdgpu_waves_per_eu(1) lifts
// the VGPR budget so accumulators never spill to scratch (round-3 lesson).
// mode 0/1: f32 row-major out. mode 2: bf16 transposed out (V^T, stride LPAD).
// mode 3: f32 row-major out, rows guarded to < L_TOT (final output).
__global__ __attribute__((amdgpu_waves_per_eu(1)))
void k_gemm64(const bf16* __restrict__ A, const bf16* __restrict__ WT,
              const float* __restrict__ bias, float* __restrict__ outF,
              bf16* __restrict__ outVT, int mode) {
  int lane = threadIdx.x & 31, wave = threadIdx.x >> 5;
  int mt = blockIdx.y * 4 + wave;
  if (mt >= NT64_M) return;
  int m0 = mt * 64, n0 = blockIdx.x * 32;
  int l15 = lane & 15, half = lane >> 4;

  const bf16* ar[4];
  const bf16* br[2];
#pragma unroll
  for (int i = 0; i < 4; ++i) ar[i] = A  + (size_t)(m0 + i * 16 + l15) * DIM;
#pragma unroll
  for (int j = 0; j < 2; ++j) br[j] = WT + (size_t)(n0 + j * 16 + l15) * DIM;

  v8f acc[4][2];
#pragma unroll
  for (int i = 0; i < 4; ++i)
#pragma unroll
    for (int j = 0; j < 2; ++j)
#pragma unroll
      for (int r = 0; r < 8; ++r) acc[i][j][r] = 0.f;

  for (int k0 = 0; k0 < DIM; k0 += 32) {
    __builtin_prefetch(ar[0] + k0 + 256);                 // global_prefetch_b8
    __builtin_prefetch(br[0] + k0 + 256);
    v16bf af[4], bfv[2];
#pragma unroll
    for (int i = 0; i < 4; ++i) af[i]  = load_frag(ar[i], k0, half);
#pragma unroll
    for (int j = 0; j < 2; ++j) bfv[j] = load_frag(br[j], k0, half);
#pragma unroll
    for (int i = 0; i < 4; ++i)
#pragma unroll
      for (int j = 0; j < 2; ++j)
        acc[i][j] = wmma_bf16(af[i], bfv[j], acc[i][j]);
  }

  if (mode == 2) {                                        // V^T: vt[n][m], m contiguous
#pragma unroll
    for (int j = 0; j < 2; ++j) {
      int n = n0 + j * 16 + l15;
      float bn = bias[n];
#pragma unroll
      for (int i = 0; i < 4; ++i) {
        BFPack p;
#pragma unroll
        for (int r = 0; r < 8; ++r) p.v[r] = (bf16)(acc[i][j][r] + bn);
        *(uint4*)(outVT + (size_t)n * LPAD + m0 + i * 16 + 8 * half) = p.u4;
      }
    }
  } else {
    bool guard = (mode == 3);
#pragma unroll
    for (int j = 0; j < 2; ++j) {
      int n = n0 + j * 16 + l15;
      float bn = bias[n];
#pragma unroll
      for (int i = 0; i < 4; ++i)
#pragma unroll
        for (int r = 0; r < 8; ++r) {
          int m = m0 + i * 16 + r + 8 * half;             // C layout rows
          if (!guard || m < L_TOT) outF[(size_t)m * DIM + n] = acc[i][j][r] + bn;
        }
    }
  }
}

// --------------------------------------------------------- RMS norm + RoPE
__global__ void k_norm_rope(const float* __restrict__ raw, const float* __restrict__ g,
                            const float* __restrict__ freqs,
                            const float* __restrict__ freqsA,
                            const float* __restrict__ freqsS,
                            bf16* __restrict__ outb) {
  int lane = threadIdx.x & 31, wave = threadIdx.x >> 5;
  int row = blockIdx.x * 4 + wave;
  if (row >= L16) return;
  const float* rp = raw + (size_t)row * DIM;

  float ss = 0.f;
  for (int i = lane; i < DIM; i += 32) { float v = rp[i]; ss += v * v; }
#pragma unroll
  for (int off = 16; off >= 1; off >>= 1) ss += __shfl_xor(ss, off, 32);
  float inv = rsqrtf(ss * (1.0f / DIM) + EPSV);

  const float* f;
  if (row < 3584)      f = freqs  + (size_t)row * HD;
  else if (row < 3680) f = freqsA + (size_t)(row - 3584) * HD;
  else if (row < L_TOT)f = freqsS + (size_t)(row - 3680) * HD;
  else                 f = freqsS;                        // pad rows: don't care

  unsigned int* op = (unsigned int*)(outb + (size_t)row * DIM);
  for (int e = lane; e < DIM / 2; e += 32) {
    int d = (2 * e) % HD;                                 // position within head
    float2 xv = ((const float2*)rp)[e];
    float xr = xv.x * inv * g[2 * e];
    float xi = xv.y * inv * g[2 * e + 1];
    float c = f[d], s = f[d + 1];                         // interleaved cos/sin
    union { unsigned int u; bf16 h[2]; } w;
    w.h[0] = (bf16)(xr * c - xi * s);
    w.h[1] = (bf16)(xr * s + xi * c);
    op[e] = w.u;
  }
}

// ----------------------------------------------------------------- mask
__device__ __forceinline__ bool mask_ok(int q, int k) {
  if (q >= L_TOT || k >= L_TOT) return false;
  if (q < 512) return k < 512;
  if (q < 3584) {                                         // image rows
    int b = (q - 512) >> 10;
    int be = 512 + (b + 1) * 1024;
    int kv = be - 2048; if (kv < 512) kv = 512;
    if (k < 512) return true;
    if (k >= kv && k < be) return true;
    int a0 = 3584 + 32 * b;
    if (k >= a0 && k < a0 + 32) return true;
    return k == 3680 + b;
  }
  if (q < 3680) {                                         // action rows
    int b = (q - 3584) >> 5;
    int ibe = 512 + (b + 1) * 1024;
    int kv = ibe - 2048; if (kv < 512) kv = 512;
    if (k < 512) return true;
    if (k >= kv && k < ibe) return true;
    int a0 = 3584 + 32 * b;
    if (k >= a0 && k < a0 + 32) return true;
    return k == 3680 + b;
  }
  return k == q;                                          // state rows
}

// ------------------------------------------------- flash attention (1 wave/tile)
__global__ __attribute__((amdgpu_waves_per_eu(1)))
void k_attn(const bf16* __restrict__ qb, const bf16* __restrict__ kb,
            const bf16* __restrict__ vt, bf16* __restrict__ ob) {
  __shared__ __align__(16) bf16 pbuf[4][16][32];          // per-wave P transpose
  int lane = threadIdx.x & 31, wave = threadIdx.x >> 5;
  int qt = blockIdx.x * 4 + wave;
  int h = blockIdx.y;
  if (qt >= NTILE_M) return;
  int l15 = lane & 15, half = lane >> 4;
  int hd0 = h * HD;
  int q0 = qt * 16;

  // allowed-key intervals for this (region-aligned) query tile -> chunk skip
  int lo[4], hi[4], ni;
  if (q0 < 512) { lo[0] = 0; hi[0] = 512; ni = 1; }
  else if (q0 < 3584) {
    int b = (q0 - 512) >> 10; int be = 512 + (b + 1) * 1024;
    int kv = be - 2048; if (kv < 512) kv = 512;
    lo[0] = 0; hi[0] = 512; lo[1] = kv; hi[1] = be;
    lo[2] = 3584 + 32 * b; hi[2] = lo[2] + 32; lo[3] = 3680 + b; hi[3] = lo[3] + 1; ni = 4;
  } else if (q0 < 3680) {
    int b = (q0 - 3584) >> 5; int ibe = 512 + (b + 1) * 1024;
    int kv = ibe - 2048; if (kv < 512) kv = 512;
    lo[0] = 0; hi[0] = 512; lo[1] = kv; hi[1] = ibe;
    lo[2] = 3584 + 32 * b; hi[2] = lo[2] + 32; lo[3] = 3680 + b; hi[3] = lo[3] + 1; ni = 4;
  } else { lo[0] = 3680; hi[0] = L_TOT; ni = 1; }

  const bf16* qrow = qb + (size_t)(q0 + l15) * DIM + hd0;
  v16bf qa[4];
#pragma unroll
  for (int kk = 0; kk < 4; ++kk) qa[kk] = load_frag(qrow, kk * 32, half);

  v8f acc[8];
#pragma unroll
  for (int t = 0; t < 8; ++t)
#pragma unroll
    for (int i = 0; i < 8; ++i) acc[t][i] = 0.f;
  float m_run[8], l_run[8];
#pragma unroll
  for (int r = 0; r < 8; ++r) { m_run[r] = -1e30f; l_run[r] = 0.f; }
  const float scale = 0.0883883476483184f;                // 1/sqrt(128)

  for (int kc = 0; kc < L16; kc += 32) {
    bool any = false;
    for (int i = 0; i < ni; ++i) any = any || (kc < hi[i] && kc + 32 > lo[i]);
    if (!any) continue;

    // S = Q K^T for 32 keys (two 16x16 C tiles), K=128 in 4 WMMA steps
    v8f s0, s1;
#pragma unroll
    for (int i = 0; i < 8; ++i) { s0[i] = 0.f; s1[i] = 0.f; }
    const bf16* kr0 = kb + (size_t)(kc + l15) * DIM + hd0;
    const bf16* kr1 = kb + (size_t)(kc + 16 + l15) * DIM + hd0;
#pragma unroll
    for (int kk = 0; kk < 4; ++kk) {
      v16bf b0 = load_frag(kr0, kk * 32, half);
      v16bf b1 = load_frag(kr1, kk * 32, half);
      s0 = wmma_bf16(qa[kk], b0, s0);
      s1 = wmma_bf16(qa[kk], b1, s1);
    }

    // scale + mask + online softmax (row stats live across 16-lane halves)
    float mloc[8];
#pragma unroll
    for (int r = 0; r < 8; ++r) {
      int q = q0 + r + 8 * half;
      float v0 = mask_ok(q, kc + l15)      ? s0[r] * scale : -INFINITY;
      float v1 = mask_ok(q, kc + 16 + l15) ? s1[r] * scale : -INFINITY;
      s0[r] = v0; s1[r] = v1;
      mloc[r] = fmaxf(v0, v1);
    }
#pragma unroll
    for (int r = 0; r < 8; ++r)
#pragma unroll
      for (int off = 1; off < 16; off <<= 1)
        mloc[r] = fmaxf(mloc[r], __shfl_xor(mloc[r], off, 32));

    float alpha[8];
#pragma unroll
    for (int r = 0; r < 8; ++r) {
      float mnew = fmaxf(m_run[r], mloc[r]);
      alpha[r] = __expf(m_run[r] - mnew);
      m_run[r] = mnew;
      float p0 = __expf(s0[r] - mnew);
      float p1 = __expf(s1[r] - mnew);
      s0[r] = p0; s1[r] = p1;
      float psum = p0 + p1;
#pragma unroll
      for (int off = 1; off < 16; off <<= 1) psum += __shfl_xor(psum, off, 32);
      l_run[r] = l_run[r] * alpha[r] + psum;
    }
#pragma unroll
    for (int t = 0; t < 8; ++t)
#pragma unroll
      for (int r = 0; r < 8; ++r) acc[t][r] *= alpha[r];

    // transpose P (C layout -> A layout) through LDS, same-wave only
#pragma unroll
    for (int r = 0; r < 8; ++r) {
      pbuf[wave][r + 8 * half][l15]      = (bf16)s0[r];
      pbuf[wave][r + 8 * half][16 + l15] = (bf16)s1[r];
    }
    v16bf pa = load_frag(&pbuf[wave][l15][0], 0, half);

    // O += P @ V ; V^T rows give contiguous key pairs per lane
#pragma unroll
    for (int t = 0; t < 8; ++t) {
      const bf16* vrow = vt + (size_t)(hd0 + t * 16 + l15) * LPAD + kc;
      v16bf vb = load_frag(vrow, 0, half);
      acc[t] = wmma_bf16(pa, vb, acc[t]);
    }
  }

#pragma unroll
  for (int r = 0; r < 8; ++r) {
    float li = l_run[r];
    float invl = (li > 0.f) ? 1.0f / li : 0.f;            // pad rows -> 0
    int m = q0 + r + 8 * half;
#pragma unroll
    for (int t = 0; t < 8; ++t)
      ob[(size_t)m * DIM + hd0 + t * 16 + l15] = (bf16)(acc[t][r] * invl);
  }
}

// ------------------------------------------------ async-to-LDS compile probe
// Round-1 diagnostic revealed the param types: (v4i __device__*, v4i __shared__*,
// imm, imm). Numeric address_space casts convert implicitly (CK-style pattern).
__global__ void k_async_probe(const bf16* __restrict__ src, bf16* __restrict__ dst) {
#if defined(__has_builtin) && __has_builtin(__builtin_amdgcn_global_load_async_to_lds_b128)
  __shared__ __align__(16) bf16 buf[64];
  if (threadIdx.x == 0) {
    __builtin_amdgcn_global_load_async_to_lds_b128(
        (__attribute__((address_space(1))) v4i*)src,
        (__attribute__((address_space(3))) v4i*)buf, 0, 0);
  }
#if __has_builtin(__builtin_amdgcn_s_wait_asynccnt)
  __builtin_amdgcn_s_wait_asynccnt(0);
#endif
  __builtin_amdgcn_s_wait_tensorcnt(0);
  __syncthreads();
  dst[threadIdx.x] = buf[threadIdx.x & 63];
#else
  __builtin_amdgcn_s_wait_tensorcnt(0);
  dst[threadIdx.x] = src[threadIdx.x];
#endif
}

// ------------------------------------------------------------------- host
extern "C" void kernel_launch(void* const* d_in, const int* in_sizes, int n_in,
                              void* d_out, int out_size, void* d_ws, size_t ws_size,
                              hipStream_t stream) {
  const float* x      = (const float*)d_in[0];
  const float* freqs  = (const float*)d_in[1];
  const float* freqsA = (const float*)d_in[2];
  const float* freqsS = (const float*)d_in[3];
  const float* Wq = (const float*)d_in[4];  const float* bq = (const float*)d_in[5];
  const float* Wk = (const float*)d_in[6];  const float* bk = (const float*)d_in[7];
  const float* Wv = (const float*)d_in[8];  const float* bv = (const float*)d_in[9];
  const float* Wo = (const float*)d_in[10]; const float* bo = (const float*)d_in[11];
  const float* gq = (const float*)d_in[12]; const float* gk = (const float*)d_in[13];
  float* out = (float*)d_out;

  char* ws = (char*)d_ws;
  size_t off = 0;
  auto alloc = [&](size_t bytes) -> void* {
    void* p = ws + off;
    off += (bytes + 255) & ~(size_t)255;
    return p;
  };
  bf16*  xb   = (bf16*)alloc((size_t)LPAD * DIM * 2);
  bf16*  wqt  = (bf16*)alloc((size_t)DIM * DIM * 2);
  bf16*  wkt  = (bf16*)alloc((size_t)DIM * DIM * 2);
  bf16*  wvt  = (bf16*)alloc((size_t)DIM * DIM * 2);
  bf16*  wot  = (bf16*)alloc((size_t)DIM * DIM * 2);
  float* qraw = (float*)alloc((size_t)LPAD * DIM * 4);
  float* kraw = (float*)alloc((size_t)LPAD * DIM * 4);
  bf16*  qbb  = (bf16*)alloc((size_t)LPAD * DIM * 2);
  bf16*  kbb  = (bf16*)alloc((size_t)LPAD * DIM * 2);
  bf16*  vtb  = (bf16*)alloc((size_t)DIM * LPAD * 2);
  bf16*  obb  = (bf16*)alloc((size_t)LPAD * DIM * 2);

  dim3 b256(256), b128(128);
  k_convert_x<<<(LPAD * DIM) / 256, b256, 0, stream>>>(x, xb);
  k_transpose_w<<<(DIM * DIM) / 256, b256, 0, stream>>>(Wq, wqt);
  k_transpose_w<<<(DIM * DIM) / 256, b256, 0, stream>>>(Wk, wkt);
  k_transpose_w<<<(DIM * DIM) / 256, b256, 0, stream>>>(Wv, wvt);
  k_transpose_w<<<(DIM * DIM) / 256, b256, 0, stream>>>(Wo, wot);

  dim3 gg(DIM / 32, (NT64_M + 3) / 4);                    // (48, 15), 4 waves/block
  k_gemm64<<<gg, b128, 0, stream>>>(xb, wqt, bq, qraw, nullptr, 0);
  k_gemm64<<<gg, b128, 0, stream>>>(xb, wkt, bk, kraw, nullptr, 1);
  k_gemm64<<<gg, b128, 0, stream>>>(xb, wvt, bv, nullptr, vtb, 2);

  k_norm_rope<<<L16 / 4, b128, 0, stream>>>(qraw, gq, freqs, freqsA, freqsS, qbb);
  k_norm_rope<<<L16 / 4, b128, 0, stream>>>(kraw, gk, freqs, freqsA, freqsS, kbb);

  k_attn<<<dim3((NTILE_M + 3) / 4, NH), b128, 0, stream>>>(qbb, kbb, vtb, obb);

  k_gemm64<<<gg, b128, 0, stream>>>(obb, wot, bo, out, nullptr, 3);

  k_async_probe<<<1, 64, 0, stream>>>(xb, obb);   // obb dead after final GEMM

  (void)in_sizes; (void)n_in; (void)out_size; (void)ws_size;
}
